// AttentionSiteDTI_18124761989715
// MI455X (gfx1250) — compile-verified, matrix operand order checked by reading, not run on hardware
//
#include <hip/hip_runtime.h>

// ---------------------------------------------------------------------------
// AttentionSiteDTI on MI455X (gfx1250, wave32).
// Dense TAGConv projections -> v_wmma_f32_16x16x32_bf16 (f32 accum).
// Sparse hops -> L2-resident fp32 atomic scatter (features fit in 192MB L2).
// ---------------------------------------------------------------------------

typedef __attribute__((ext_vector_type(16))) __bf16 v16bf;
typedef __attribute__((ext_vector_type(8)))  __bf16 v8bf;
typedef __attribute__((ext_vector_type(8)))  float  v8f;

static inline int cdiv(int a, int b) { return (a + b - 1) / b; }

__device__ __forceinline__ unsigned short f32_to_bf16_bits(float f) {
  unsigned u = __float_as_uint(f);
  u += 0x7FFFu + ((u >> 16) & 1u);   // round-to-nearest-even
  return (unsigned short)(u >> 16);
}

// ------------------------------ utility kernels ----------------------------

__global__ void zero_u32(unsigned* __restrict__ p, int count) {
  int i = blockIdx.x * blockDim.x + threadIdx.x;
  if (i < count) p[i] = 0u;
}

__global__ void deg_count(const int* __restrict__ dst, float* __restrict__ deg, int nE) {
  int i = blockIdx.x * blockDim.x + threadIdx.x;
  if (i < nE) atomicAdd(&deg[dst[i]], 1.0f);
}

__global__ void make_norm(const float* __restrict__ deg, float* __restrict__ norm, int n) {
  int i = blockIdx.x * blockDim.x + threadIdx.x;
  if (i < n) {
    float d = deg[i];
    norm[i] = rsqrtf(d > 0.0f ? d : 1.0f);
  }
}

// hop 0: raw features -> bf16 columns [0, d)
__global__ void cast_hop0(const float* __restrict__ x, unsigned short* __restrict__ hops,
                          int n, int d, int Kpad) {
  int idx = blockIdx.x * blockDim.x + threadIdx.x;
  if (idx >= n * d) return;
  int i = idx / d, f = idx - i * d;
  hops[(long long)i * Kpad + f] = f32_to_bf16_bits(x[idx]);
}

// acc[dst] += h[src] * norm[src]   (one wave per edge; lanes stride features)
__global__ void edge_scatter(const float* __restrict__ h, const float* __restrict__ norm,
                             const int* __restrict__ src, const int* __restrict__ dst,
                             float* __restrict__ acc, int nE, int d) {
  int wave = (blockIdx.x * blockDim.x + threadIdx.x) >> 5;
  int lane = threadIdx.x & 31;
  if (wave >= nE) return;
  int s = src[wave], t = dst[wave];
  float ns = norm[s];
  const float* hs = h + (long long)s * d;
  float* at = acc + (long long)t * d;
  for (int f = lane; f < d; f += 32) atomicAdd(at + f, hs[f] * ns);
}

// h_new = acc * norm ; also write bf16 into hops concat block
__global__ void finish_hop(const float* __restrict__ acc, const float* __restrict__ norm,
                           float* __restrict__ h, unsigned short* __restrict__ hops,
                           int n, int d, int colbase, int Kpad) {
  int idx = blockIdx.x * blockDim.x + threadIdx.x;
  if (idx >= n * d) return;
  int i = idx / d, f = idx - i * d;
  float v = acc[idx] * norm[i];
  h[idx] = v;
  hops[(long long)i * Kpad + colbase + f] = f32_to_bf16_bits(v);
}

// W [Kdim, N] f32 -> Wt [Npad, Kpad] bf16 (transposed, zero-padded)
__global__ void conv_weight(const float* __restrict__ W, unsigned short* __restrict__ Wt,
                            int Kdim, int N, int Kpad, int Npad) {
  int idx = blockIdx.x * blockDim.x + threadIdx.x;
  if (idx >= Npad * Kpad) return;
  int nn = idx / Kpad, kk = idx - nn * Kpad;
  float v = (nn < N && kk < Kdim) ? W[(long long)kk * N + nn] : 0.0f;
  Wt[idx] = f32_to_bf16_bits(v);
}

__global__ void pad_bias(const float* __restrict__ b, float* __restrict__ bp, int N, int Npad) {
  int i = threadIdx.x;
  if (i < Npad) bp[i] = (i < N) ? b[i] : 0.0f;
}

// ------------------------- WMMA GEMM + relu + l2norm -----------------------
// Y[n, dout] = relu(Hops[n, Kpad](bf16) @ W[Kpad, Npad](bf16) + bias), row l2-normalized.
// One wave per 16-row M-tile; NT 16-col N-tiles unrolled (v_wmma per tile/K-step).
template <int NT>
__global__ __launch_bounds__(256) void gnn_gemm(
    const unsigned short* __restrict__ hops_u, const unsigned short* __restrict__ Wt_u,
    const float* __restrict__ bpad, float* __restrict__ out,
    int n, int Kpad, int dout) {
  __shared__ unsigned short sW[64 * 384];  // Npad * Kpad max = 49152 B
  int tid = threadIdx.x;
  int words = (NT * 16 * Kpad) >> 1;
  const unsigned* Wsrc = (const unsigned*)Wt_u;
  unsigned* Wdst = (unsigned*)sW;
  for (int i = tid; i < words; i += 256) Wdst[i] = Wsrc[i];
  __syncthreads();

  int wave = tid >> 5, lane = tid & 31;
  int tileM = blockIdx.x * 8 + wave;
  if (tileM * 16 >= n) return;          // wave-uniform exit, EXEC stays all-ones
  int col = lane & 15, khalf = lane >> 4;
  int mrow = tileM * 16 + col;
  if (mrow >= n) mrow = n - 1;          // duplicate row; stores are guarded
  const __bf16* arow = (const __bf16*)hops_u + (long long)mrow * Kpad + khalf * 8;
  const __bf16* sWb  = (const __bf16*)sW;

  v8f acc[NT];
  v8f z = {0.f, 0.f, 0.f, 0.f, 0.f, 0.f, 0.f, 0.f};
#pragma unroll
  for (int t = 0; t < NT; ++t) acc[t] = z;

  for (int kb = 0; kb < Kpad; kb += 32) {
    // A fragment (16x32 bf16): lane holds row mrow, two 8-elem contiguous chunks
    v8bf alo = *(const v8bf*)(arow + kb);
    v8bf ahi = *(const v8bf*)(arow + kb + 16);
    v16bf a;
#pragma unroll
    for (int i = 0; i < 8; ++i) { a[i] = alo[i]; a[i + 8] = ahi[i]; }
#pragma unroll
    for (int t = 0; t < NT; ++t) {
      // B fragment (32x16 bf16) from transposed weights in LDS (same chunking)
      const __bf16* brow = sWb + (long long)(t * 16 + col) * Kpad + khalf * 8 + kb;
      v8bf blo = *(const v8bf*)(brow);
      v8bf bhi = *(const v8bf*)(brow + 16);
      v16bf b;
#pragma unroll
      for (int i = 0; i < 8; ++i) { b[i] = blo[i]; b[i + 8] = bhi[i]; }
      acc[t] = __builtin_amdgcn_wmma_f32_16x16x32_bf16(
          false, a, false, b, (short)0, acc[t], false, false);
    }
  }

  // bias + relu + row sum-of-squares (C/D layout: vgpr i -> row i + 8*khalf, lane -> col)
  float vals[NT][8];
  float ss[8];
#pragma unroll
  for (int i = 0; i < 8; ++i) ss[i] = 0.0f;
#pragma unroll
  for (int t = 0; t < NT; ++t) {
    float bb = bpad[t * 16 + col];
#pragma unroll
    for (int i = 0; i < 8; ++i) {
      float v = acc[t][i] + bb;
      v = v > 0.0f ? v : 0.0f;
      vals[t][i] = v;
      ss[i] += v * v;
    }
  }
#pragma unroll
  for (int i = 0; i < 8; ++i) {
    float s = ss[i];
    for (int m = 1; m < 16; m <<= 1) s += __shfl_xor(s, m, 32);  // within 16-lane half
    ss[i] = 1.0f / fmaxf(sqrtf(s), 1e-12f);
  }
  int rbase = tileM * 16 + khalf * 8;
#pragma unroll
  for (int t = 0; t < NT; ++t) {
    int c = t * 16 + col;
    if (c >= dout) continue;
#pragma unroll
    for (int i = 0; i < 8; ++i) {
      int r = rbase + i;
      if (r < n) out[(long long)r * dout + c] = vals[t][i] * ss[i];
    }
  }
}

// segment-max pooling (values >= 0 after relu+l2norm -> uint compare is monotone)
__global__ void pool_max(const float* __restrict__ y, const int* __restrict__ gid,
                         float* __restrict__ pool, int n, int d) {
  int idx = blockIdx.x * blockDim.x + threadIdx.x;
  if (idx >= n * d) return;
  int i = idx / d, f = idx - i * d;
  atomicMax((unsigned*)&pool[gid[i] * d + f], __float_as_uint(y[idx]));
}

// ------------------------------ attention path -----------------------------

__global__ void build_seq(const float* __restrict__ lig_pool,
                          const float* __restrict__ prot_pool, float* __restrict__ seq) {
  int idx = blockIdx.x * blockDim.x + threadIdx.x;
  if (idx >= 130 * 45) return;
  int r = idx / 45, c = idx - r * 45;
  float v = 0.0f;
  if (r == 0) v = lig_pool[c];
  else if (r <= 64) v = prot_pool[(r - 1) * 45 + c];
  seq[idx] = v;
}

__global__ void small_gemm(const float* __restrict__ X, const float* __restrict__ W,
                           const float* __restrict__ b, float* __restrict__ Y,
                           int M, int K, int N) {
  int idx = blockIdx.x * blockDim.x + threadIdx.x;
  if (idx >= M * N) return;
  int r = idx / N, c = idx - r * N;
  float s = 0.0f;
  for (int k = 0; k < K; ++k) s += X[r * K + k] * W[k * N + c];
  Y[idx] = s + b[c];
}

__global__ __launch_bounds__(256) void attn_softmax_av(const float* __restrict__ qkv,
                                                       float* __restrict__ out) {
  __shared__ float p[132];
  __shared__ float ssum;
  int i = blockIdx.x, tid = threadIdx.x;
  if (tid < 130) {
    int j = tid;
    bool allowed = ((i == j && i < 65) || i == 64 || j == 64) && !(i == 64 && j == 64);
    float s = -1e9f;
    if (allowed) {
      float a = 0.0f;
      const float* qi = qkv + i * 135;
      const float* kj = qkv + j * 135 + 45;
      for (int d = 0; d < 45; ++d) a += qi[d] * kj[d];
      s = a * 0.14907119849998599f;  // 45^-0.5
    }
    p[j] = s;
  }
  __syncthreads();
  if (tid == 0) {
    float m = -1e30f;
    for (int j = 0; j < 130; ++j) m = fmaxf(m, p[j]);
    float sum = 0.0f;
    for (int j = 0; j < 130; ++j) { float e = expf(p[j] - m); p[j] = e; sum += e; }
    ssum = sum;
  }
  __syncthreads();
  if (tid < 45) {
    float inv = 1.0f / ssum;
    float a = 0.0f;
    for (int j = 0; j < 130; ++j) a += p[j] * qkv[j * 135 + 90 + tid];
    out[i * 45 + tid] = a * inv;
  }
}

// batch-1 GEMV: thread-per-output-column, coalesced weight rows. act: 1=relu, 2=sigmoid
__global__ void gemv_act(const float* __restrict__ x, const float* __restrict__ W,
                         const float* __restrict__ b, float* __restrict__ out,
                         int K, int N, int act) {
  int nidx = blockIdx.x * blockDim.x + threadIdx.x;
  if (nidx >= N) return;
  float s = 0.0f;
  for (int k = 0; k < K; ++k) s += x[k] * W[(long long)k * N + nidx];
  s += b[nidx];
  if (act == 1) s = fmaxf(s, 0.0f);
  else if (act == 2) s = 1.0f / (1.0f + expf(-s));
  out[nidx] = s;
}

// ------------------------------ host orchestration -------------------------

static void run_gnn_layer(const float* x, const int* src, const int* dst,
                          const float* norm, int n, int nE,
                          const float* Wf, const float* bf, int din, int dout,
                          float* h, float* acc, unsigned short* hops,
                          unsigned short* Wt, float* bpad, float* y,
                          hipStream_t s) {
  int Kdim = 5 * din;
  int Kpad = (Kdim + 31) & ~31;
  int Npad = (dout + 15) & ~15;
  cast_hop0<<<cdiv(n * din, 256), 256, 0, s>>>(x, hops, n, din, Kpad);
  const float* hprev = x;
  for (int k = 1; k <= 4; ++k) {
    zero_u32<<<cdiv(n * din, 256), 256, 0, s>>>((unsigned*)acc, n * din);
    edge_scatter<<<cdiv(nE, 8), 256, 0, s>>>(hprev, norm, src, dst, acc, nE, din);
    finish_hop<<<cdiv(n * din, 256), 256, 0, s>>>(acc, norm, h, hops, n, din, k * din, Kpad);
    hprev = h;
  }
  conv_weight<<<cdiv(Npad * Kpad, 256), 256, 0, s>>>(Wf, Wt, Kdim, dout, Kpad, Npad);
  pad_bias<<<1, 64, 0, s>>>(bf, bpad, dout, Npad);
  int grid = cdiv(n, 128);  // 8 waves * 16 rows per block
  if (Npad == 64) gnn_gemm<4><<<grid, 256, 0, s>>>(hops, Wt, bpad, y, n, Kpad, dout);
  else            gnn_gemm<3><<<grid, 256, 0, s>>>(hops, Wt, bpad, y, n, Kpad, dout);
}

extern "C" void kernel_launch(void* const* d_in, const int* in_sizes, int n_in,
                              void* d_out, int out_size, void* d_ws, size_t ws_size,
                              hipStream_t stream) {
  const float* prot_x = (const float*)d_in[0];
  const float* lig_x  = (const float*)d_in[1];
  const float* pW[3] = {(const float*)d_in[2], (const float*)d_in[4], (const float*)d_in[6]};
  const float* pB[3] = {(const float*)d_in[3], (const float*)d_in[5], (const float*)d_in[7]};
  const float* lW[4] = {(const float*)d_in[8], (const float*)d_in[10], (const float*)d_in[12], (const float*)d_in[14]};
  const float* lB[4] = {(const float*)d_in[9], (const float*)d_in[11], (const float*)d_in[13], (const float*)d_in[15]};
  const float* qkv_w  = (const float*)d_in[16];
  const float* qkv_b  = (const float*)d_in[17];
  const float* proj_w = (const float*)d_in[18];
  const float* proj_b = (const float*)d_in[19];
  const float* hw[4] = {(const float*)d_in[20], (const float*)d_in[22], (const float*)d_in[24], (const float*)d_in[26]};
  const float* hb[4] = {(const float*)d_in[21], (const float*)d_in[23], (const float*)d_in[25], (const float*)d_in[27]};
  const int* prot_src = (const int*)d_in[28];
  const int* prot_dst = (const int*)d_in[29];
  const int* prot_gid = (const int*)d_in[30];
  const int* lig_src  = (const int*)d_in[31];
  const int* lig_dst  = (const int*)d_in[32];
  const int* lig_gid  = (const int*)d_in[33];

  const int NP = 100000, NL = 5000, EP = 1600000, EL = 80000;

  // workspace carve-out (256B aligned blocks)
  char* base = (char*)d_ws;
  size_t off = 0;
  auto take = [&](size_t bytes) -> char* {
    char* p = base + off;
    off += (bytes + 255) & ~(size_t)255;
    return p;
  };
  float*          norm = (float*)take((size_t)NP * 4);
  float*          deg  = (float*)take((size_t)NP * 4);
  float*          hbuf = (float*)take((size_t)NP * 74 * 4);
  float*          accb = (float*)take((size_t)NP * 74 * 4);
  unsigned short* hops = (unsigned short*)take((size_t)NP * 384 * 2);
  float*          y0   = (float*)take((size_t)NP * 64 * 4);
  float*          y1   = (float*)take((size_t)NP * 64 * 4);
  unsigned short* Wt   = (unsigned short*)take((size_t)64 * 384 * 2);
  float*          bpad = (float*)take(64 * 4);
  float*     prot_pool = (float*)take(64 * 45 * 4);
  float*      lig_pool = (float*)take(45 * 4);
  float*           seq = (float*)take(130 * 45 * 4);
  float*          qkvb = (float*)take(130 * 135 * 4);
  float*         attno = (float*)take(130 * 45 * 4);
  float*         proje = (float*)take(130 * 45 * 4);
  float*           hd1 = (float*)take(2000 * 4);
  float*           hd2 = (float*)take(1000 * 4);
  float*           hd3 = (float*)take(500 * 4);
  (void)ws_size; (void)in_sizes; (void)n_in; (void)out_size;

  hipStream_t s = stream;

  // ---------------- protein GNN ----------------
  zero_u32<<<cdiv(NP, 256), 256, 0, s>>>((unsigned*)deg, NP);
  deg_count<<<cdiv(EP, 256), 256, 0, s>>>(prot_dst, deg, EP);
  make_norm<<<cdiv(NP, 256), 256, 0, s>>>(deg, norm, NP);
  zero_u32<<<cdiv(NP * 192, 256), 256, 0, s>>>((unsigned*)hops, NP * 192);
  run_gnn_layer(prot_x, prot_src, prot_dst, norm, NP, EP, pW[0], pB[0], 74, 50,
                hbuf, accb, hops, Wt, bpad, y0, s);
  run_gnn_layer(y0, prot_src, prot_dst, norm, NP, EP, pW[1], pB[1], 50, 45,
                hbuf, accb, hops, Wt, bpad, y1, s);
  run_gnn_layer(y1, prot_src, prot_dst, norm, NP, EP, pW[2], pB[2], 45, 45,
                hbuf, accb, hops, Wt, bpad, y0, s);
  zero_u32<<<cdiv(64 * 45, 256), 256, 0, s>>>((unsigned*)prot_pool, 64 * 45);
  pool_max<<<cdiv(NP * 45, 256), 256, 0, s>>>(y0, prot_gid, prot_pool, NP, 45);

  // ---------------- ligand GNN ----------------
  zero_u32<<<cdiv(NL, 256), 256, 0, s>>>((unsigned*)deg, NL);
  deg_count<<<cdiv(EL, 256), 256, 0, s>>>(lig_dst, deg, EL);
  make_norm<<<cdiv(NL, 256), 256, 0, s>>>(deg, norm, NL);
  zero_u32<<<cdiv(NL * 192, 256), 256, 0, s>>>((unsigned*)hops, NL * 192);
  run_gnn_layer(lig_x, lig_src, lig_dst, norm, NL, EL, lW[0], lB[0], 74, 50,
                hbuf, accb, hops, Wt, bpad, y0, s);
  run_gnn_layer(y0, lig_src, lig_dst, norm, NL, EL, lW[1], lB[1], 50, 45,
                hbuf, accb, hops, Wt, bpad, y1, s);
  run_gnn_layer(y1, lig_src, lig_dst, norm, NL, EL, lW[2], lB[2], 45, 45,
                hbuf, accb, hops, Wt, bpad, y0, s);
  run_gnn_layer(y0, lig_src, lig_dst, norm, NL, EL, lW[3], lB[3], 45, 45,
                hbuf, accb, hops, Wt, bpad, y1, s);
  zero_u32<<<1, 64, 0, s>>>((unsigned*)lig_pool, 45);
  pool_max<<<cdiv(NL * 45, 256), 256, 0, s>>>(y1, lig_gid, lig_pool, NL, 45);

  // ---------------- attention ----------------
  build_seq<<<cdiv(130 * 45, 256), 256, 0, s>>>(lig_pool, prot_pool, seq);
  small_gemm<<<cdiv(130 * 135, 256), 256, 0, s>>>(seq, qkv_w, qkv_b, qkvb, 130, 45, 135);
  attn_softmax_av<<<130, 256, 0, s>>>(qkvb, attno);
  small_gemm<<<cdiv(130 * 45, 256), 256, 0, s>>>(attno, proj_w, proj_b, proje, 130, 45, 45);

  // ---------------- head MLP ----------------
  gemv_act<<<cdiv(2000, 256), 256, 0, s>>>(proje, hw[0], hb[0], hd1, 5850, 2000, 1);
  gemv_act<<<cdiv(1000, 256), 256, 0, s>>>(hd1, hw[1], hb[1], hd2, 2000, 1000, 1);
  gemv_act<<<cdiv(500, 256), 256, 0, s>>>(hd2, hw[2], hb[2], hd3, 1000, 500, 1);
  gemv_act<<<1, 256, 0, s>>>(hd3, hw[3], hb[3], (float*)d_out, 500, 1, 2);
}
